// GRURouterCell_82231443849805
// MI455X (gfx1250) — matrix operand order repaired and checked by reading, not compiled, not run on hardware
//
#include <hip/hip_runtime.h>

// ---------------------------------------------------------------------------
// GRU cell, B=32768, I=H=1024, fused dual-GEMM (x@Wi.T, h@Wh.T) + gating.
// bf16 WMMA (v_wmma_f32_16x16x32_bf16) with fp32 accumulation.
// Tiling: block 256 thr (8 waves), tile 128(M) x 32(N), K-step 32.
// Each wave: 32x16 output patch, 6 gate accumulators (96 VGPRs).
// LDS double-buffered, rows padded to 40 bf16 (80B) for conflict-free b128.
// bf16 operands staged global->LDS with GLOBAL_LOAD_ASYNC_TO_LDS_B128
// (ASYNCcnt), removing the loadcnt/ds_store ladder and staging VGPRs.
// ---------------------------------------------------------------------------

#define BDIM 256
constexpr int BM   = 128;
constexpr int BN   = 32;
constexpr int KT   = 32;
constexpr int KDIM = 1024;
constexpr int HD   = 1024;
constexpr int NK   = KDIM / KT;        // 32 K-iterations
constexpr int LDK  = 40;               // padded row stride in bf16 elems (80B)
constexpr int XOFF = 0;
constexpr int XSZ  = BM * LDK;         // 5120
constexpr int HOFF = XOFF + XSZ;       // 5120
constexpr int WOFF = HOFF + XSZ;       // 10240
constexpr int WSZ  = BN * LDK;         // 1280 per gate
constexpr int STAGE = WOFF + 6 * WSZ;  // 17920 u16 = 35840 B per stage

typedef __attribute__((ext_vector_type(4)))  unsigned int   v4u;
typedef __attribute__((ext_vector_type(4)))  int            v4i;
typedef __attribute__((ext_vector_type(4)))  unsigned short v4h;
typedef __attribute__((ext_vector_type(4)))  float          v4f;
typedef __attribute__((ext_vector_type(8)))  float          v8f;
typedef __attribute__((ext_vector_type(16))) __bf16         v16bf;

union Frag { v4u u[2]; v16bf v; };

struct Params {
  const float* x;                  // [32768,1024] fp32
  const float* h;                  // [32768,1024] fp32
  const float* w[6];               // ir,hr,iz,hz,in,hn  [1024,1024] fp32
  const float* b[6];               // matching biases [1024]
  const unsigned short* xb;        // bf16 mirrors in workspace (mode-dependent)
  const unsigned short* hb;
  const unsigned short* wb;        // [6][1024][1024] bf16
  float* out;
};

__device__ __forceinline__ unsigned short f2bf(float f) {
  unsigned u = __float_as_uint(f);
  return (unsigned short)((u + 0x7FFFu + ((u >> 16) & 1u)) >> 16);  // RNE
}
__device__ __forceinline__ float sigm(float v) { return 1.0f / (1.0f + __expf(-v)); }

// ---- CDNA5 async global->LDS (ASYNCcnt) ------------------------------------
// Builtin signature (probed from diagnostics): arg0 = AS(1) v4i* (global),
// arg1 = AS(3) v4i* (LDS), arg2/arg3 = imm offset / imm cpol.
typedef __attribute__((address_space(1))) v4i* as1_v4i;
typedef __attribute__((address_space(3))) v4i* as3_v4i;

__device__ __forceinline__ void async_copy_b128(const void* g, const void* l) {
  // AS3 pointer == 32-bit LDS byte offset (flat addr truncation rule).
#if __has_builtin(__builtin_amdgcn_global_load_async_to_lds_b128)
  __builtin_amdgcn_global_load_async_to_lds_b128(
      (as1_v4i)(unsigned long long)g,
      (as3_v4i)(unsigned long long)l, 0, 0);
#else
  asm volatile("global_load_async_to_lds_b128 %0, %1, off"
               :: "v"((unsigned)(unsigned long long)l),
                  "v"((unsigned long long)g)
               : "memory");
#endif
}
__device__ __forceinline__ void wait_async0() {
#if __has_builtin(__builtin_amdgcn_s_wait_asynccnt)
  __builtin_amdgcn_s_wait_asynccnt(0);
#else
  asm volatile("s_wait_asynccnt 0" ::: "memory");
#endif
}

// A fragment (16x32 bf16): u[0] -> VGPR0-3 = K base..base+7,
// u[1] -> VGPR4-7 = K base+16..base+23, base = kh*8 (ISA layout).
__device__ __forceinline__ v16bf ldA(const unsigned short* s, int off, int row, int kh) {
  Frag f;
  const int o = off + row * LDK + kh * 8;
  f.u[0] = *(const v4u*)(s + o);
  f.u[1] = *(const v4u*)(s + o + 16);
  return f.v;
}
// B fragment (32x16 bf16): lane = column; K base..base+7, base+16..base+23, base = kh*8
__device__ __forceinline__ v16bf ldB(const unsigned short* s, int off, int row, int kh) {
  Frag f;
  const int o = off + row * LDK + kh * 8;
  f.u[0] = *(const v4u*)(s + o);
  f.u[1] = *(const v4u*)(s + o + 16);
  return f.v;
}

// MODE 0: all operands fp32 from global (convert on the fly, VGPR staging)
// MODE 1: weights bf16 in ws (async->LDS), x/h fp32 staged + converted
// MODE 2: x, h, weights all bf16 in ws; all staging via async->LDS
template<int MODE>
__global__ __launch_bounds__(BDIM)
void gru_wmma_kernel(Params p) {
  __shared__ unsigned short smem[2 * STAGE];

  const int tid  = threadIdx.x;
  const int lane = tid & 31;
  const int wid  = tid >> 5;
  const int wm   = wid & 3;        // 4 M-groups of 32 rows
  const int wn   = wid >> 2;       // 2 N-groups of 16 cols
  const int gm   = blockIdx.y * BM;
  const int gn   = blockIdx.x * BN;

  v8f acc[6][2] = {};              // [gate][m-subtile], fp32 accum

  v4f rxf[4], rhf[4], rwf[6];      // fp32 staging regs (MODE<2 only)

  // ---- async bf16 staging: global -> LDS, no VGPR round-trip ----
  auto issueAsync = [&](int st, int kt) {
    const int k0 = kt * KT;
    unsigned short* sb = smem + st * STAGE;
    if constexpr (MODE == 2) {
#pragma unroll
      for (int j = 0; j < 2; ++j) {
        const int c = tid + j * BDIM;                    // 0..511
        const int r = c >> 2, col = (c & 3) << 3;
        async_copy_b128(p.xb + (size_t)(gm + r) * KDIM + k0 + col,
                        sb + XOFF + r * LDK + col);
        async_copy_b128(p.hb + (size_t)(gm + r) * KDIM + k0 + col,
                        sb + HOFF + r * LDK + col);
      }
    }
    if constexpr (MODE >= 1) {
#pragma unroll
      for (int j = 0; j < 3; ++j) {
        const int c = tid + j * BDIM;                    // 0..767
        const int g = c >> 7, r = (c >> 2) & 31, col = (c & 3) << 3;
        async_copy_b128(p.wb + ((size_t)g << 20) + (size_t)(gn + r) * KDIM + k0 + col,
                        sb + WOFF + g * WSZ + r * LDK + col);
      }
    }
  };

  // ---- fp32 staging (needs on-the-fly conversion) ----
  auto loadRegs = [&](int kt) {
    const int k0 = kt * KT;
    if constexpr (MODE <= 1) {
#pragma unroll
      for (int j = 0; j < 4; ++j) {
        const int c = tid + j * BDIM;                    // 0..1023
        const int r = c >> 3, col = (c & 7) << 2;
        rxf[j] = *(const v4f*)(p.x + (size_t)(gm + r) * KDIM + k0 + col);
        rhf[j] = *(const v4f*)(p.h + (size_t)(gm + r) * KDIM + k0 + col);
      }
    }
    if constexpr (MODE == 0) {
#pragma unroll
      for (int j = 0; j < 6; ++j) {
        const int c = tid + j * BDIM;                    // 0..1535
        const int g = c >> 8, r = (c >> 3) & 31, col = (c & 7) << 2;
        rwf[j] = *(const v4f*)(p.w[g] + (size_t)(gn + r) * KDIM + k0 + col);
      }
    }
  };

  auto storeLds = [&](int st) {
    unsigned short* sb = smem + st * STAGE;
    if constexpr (MODE <= 1) {
#pragma unroll
      for (int j = 0; j < 4; ++j) {
        const int c = tid + j * BDIM;
        const int r = c >> 3, col = (c & 7) << 2;
        v4h xo, ho;
#pragma unroll
        for (int k = 0; k < 4; ++k) { xo[k] = f2bf(rxf[j][k]); ho[k] = f2bf(rhf[j][k]); }
        *(v4h*)(sb + XOFF + r * LDK + col) = xo;
        *(v4h*)(sb + HOFF + r * LDK + col) = ho;
      }
    }
    if constexpr (MODE == 0) {
#pragma unroll
      for (int j = 0; j < 6; ++j) {
        const int c = tid + j * BDIM;
        const int g = c >> 8, r = (c >> 3) & 31, col = (c & 7) << 2;
        v4h wo;
#pragma unroll
        for (int k = 0; k < 4; ++k) wo[k] = f2bf(rwf[j][k]);
        *(v4h*)(sb + WOFF + g * WSZ + r * LDK + col) = wo;
      }
    }
  };

  // ---- prologue: fill stage 0 ----
  if constexpr (MODE >= 1) issueAsync(0, 0);
  if constexpr (MODE <= 1) { loadRegs(0); storeLds(0); }
  if constexpr (MODE >= 1) wait_async0();
  __syncthreads();

  const int ra = lane & 15;
  const int kh = lane >> 4;

  for (int kt = 0; kt < NK; ++kt) {
    const bool hasNext = (kt + 1 < NK);
    if (hasNext) {
      if constexpr (MODE >= 1) issueAsync((kt + 1) & 1, kt + 1);  // background DMA
      if constexpr (MODE <= 1) loadRegs(kt + 1);
    }

    const unsigned short* sb = smem + (kt & 1) * STAGE;
    v16bf ax0 = ldA(sb, XOFF, wm * 32 + ra,      kh);
    v16bf ax1 = ldA(sb, XOFF, wm * 32 + 16 + ra, kh);
    v16bf ah0 = ldA(sb, HOFF, wm * 32 + ra,      kh);
    v16bf ah1 = ldA(sb, HOFF, wm * 32 + 16 + ra, kh);

#pragma unroll
    for (int g = 0; g < 6; ++g) {            // even gates use x, odd use h
      v16bf bw = ldB(sb, WOFF + g * WSZ, wn * 16 + ra, kh);
      v16bf A0 = (g & 1) ? ah0 : ax0;
      v16bf A1 = (g & 1) ? ah1 : ax1;
      acc[g][0] = __builtin_amdgcn_wmma_f32_16x16x32_bf16(
          false, A0, false, bw, (short)0, acc[g][0], false, false);
      acc[g][1] = __builtin_amdgcn_wmma_f32_16x16x32_bf16(
          false, A1, false, bw, (short)0, acc[g][1], false, false);
    }

    if (hasNext) {
      if constexpr (MODE <= 1) storeLds((kt + 1) & 1);
      if constexpr (MODE >= 1) wait_async0();     // DMA complete before barrier
    }
    __syncthreads();
  }

  // ---- fused gating epilogue ----
  // C/D layout: VGPR j -> M = j + 8*(lane>=16), N = lane&15
  const int col = gn + wn * 16 + ra;
  float bb[6];
#pragma unroll
  for (int g = 0; g < 6; ++g) bb[g] = p.b[g][col];
  const int m0 = gm + wm * 32 + (kh << 3);
#pragma unroll
  for (int mi = 0; mi < 2; ++mi) {
#pragma unroll
    for (int j = 0; j < 8; ++j) {
      const int    M   = m0 + mi * 16 + j;
      const size_t idx = (size_t)M * HD + col;
      const float  hv  = p.h[idx];
      const float r = sigm(acc[0][mi][j] + acc[1][mi][j] + bb[0] + bb[1]);   // ir+hr
      const float z = sigm(acc[2][mi][j] + acc[3][mi][j] + bb[2] + bb[3]);   // iz+hz
      const float n = tanhf(acc[4][mi][j] + bb[4] + r * (acc[5][mi][j] + bb[5]));
      p.out[idx] = (1.0f - z) * n + z * hv;
    }
  }
}

__global__ __launch_bounds__(256)
void cvt_f32_bf16_kernel(const float* __restrict__ src,
                         unsigned short* __restrict__ dst, int n4) {
  const int i = blockIdx.x * 256 + threadIdx.x;
  if (i < n4) {
    const v4f f = ((const v4f*)src)[i];
    v4h o;
#pragma unroll
    for (int k = 0; k < 4; ++k) o[k] = f2bf(f[k]);
    ((v4h*)dst)[i] = o;
  }
}

extern "C" void kernel_launch(void* const* d_in, const int* in_sizes, int n_in,
                              void* d_out, int out_size, void* d_ws, size_t ws_size,
                              hipStream_t stream) {
  (void)in_sizes; (void)n_in; (void)out_size;

  Params p;
  p.x = (const float*)d_in[0];
  p.h = (const float*)d_in[1];
  for (int g = 0; g < 6; ++g) {
    p.w[g] = (const float*)d_in[2 + g];   // ir, hr, iz, hz, in, hn
    p.b[g] = (const float*)d_in[8 + g];   // matching bias order
  }
  p.out = (float*)d_out;

  const size_t W_ELEMS = 6ull * 1024 * 1024;    // 6,291,456
  const size_t X_ELEMS = 32768ull * 1024;       // 33,554,432
  const size_t W_BYTES = W_ELEMS * 2;           // 12,582,912
  const size_t X_BYTES = X_ELEMS * 2;           // 67,108,864

  int mode = 0;
  if (ws_size >= W_BYTES + 2 * X_BYTES) mode = 2;       // everything bf16 (fits L2)
  else if (ws_size >= W_BYTES)          mode = 1;       // weights bf16 only

  unsigned short* wb = (unsigned short*)d_ws;
  unsigned short* xb = wb + W_ELEMS;
  unsigned short* hb = xb + X_ELEMS;
  p.wb = wb; p.xb = xb; p.hb = hb;

  if (mode >= 1) {
    const int n4 = 1024 * 1024 / 4;                     // per weight matrix
    for (int g = 0; g < 6; ++g)
      cvt_f32_bf16_kernel<<<n4 / 256, 256, 0, stream>>>(p.w[g], wb + (size_t)g * 1024 * 1024, n4);
  }
  if (mode == 2) {
    const int n4 = (int)(X_ELEMS / 4);                  // 8,388,608
    cvt_f32_bf16_kernel<<<n4 / 256, 256, 0, stream>>>(p.x, xb, n4);
    cvt_f32_bf16_kernel<<<n4 / 256, 256, 0, stream>>>(p.h, hb, n4);
  }

  dim3 grid(1024 / BN, 32768 / BM);   // (32 N-tiles, 256 M-tiles)
  dim3 block(BDIM);
  if (mode == 2)      gru_wmma_kernel<2><<<grid, block, 0, stream>>>(p);
  else if (mode == 1) gru_wmma_kernel<1><<<grid, block, 0, stream>>>(p);
  else                gru_wmma_kernel<0><<<grid, block, 0, stream>>>(p);
}
